// MultiHeadLatentAttention_70978629533966
// MI455X (gfx1250) — matrix operand order repaired and checked by reading, not compiled, not run on hardware
//
#include <hip/hip_runtime.h>
#include <hip/hip_bf16.h>

typedef __attribute__((ext_vector_type(16))) _Float16 v16h;
typedef __attribute__((ext_vector_type(8)))  _Float16 v8h;
typedef __attribute__((ext_vector_type(8)))  float    v8f;

#define B_  2
#define S_  2048
#define D_  2048
#define H_  16
#define DH_ 128
#define M_  (B_*S_)          // 4096 total rows
#define CLAMP_ 10.0f

__device__ __forceinline__ float clamp10(float v) {
    return fminf(fmaxf(v, -CLAMP_), CLAMP_);
}

// ---------------------------------------------------------------------------
// CDNA5 async memory->LDS copy (tracked by ASYNCcnt), 16 bytes per lane.
// global_load_async_to_lds_b128: VDST = LDS byte address VGPR, VADDR = 64-bit
// global address (GV mode, saddr = off).
// ---------------------------------------------------------------------------
__device__ __forceinline__ void async_copy_b128(const _Float16* g, _Float16* l) {
    asm volatile("global_load_async_to_lds_b128 %0, %1, off"
                 :
                 : "v"((__attribute__((address_space(3))) _Float16*)l),
                   "v"((__attribute__((address_space(1))) const _Float16*)g)
                 : "memory");
}
__device__ __forceinline__ void wait_async0() {
    asm volatile("s_wait_asynccnt 0x0" ::: "memory");
}

// ---------------------------------------------------------------------------
// elementwise f32 -> f16 conversion (optionally clamped to +-10)
// ---------------------------------------------------------------------------
__global__ void cvt_f32_f16(const float* __restrict__ in, _Float16* __restrict__ out,
                            int n, int doClamp) {
    int i = blockIdx.x * blockDim.x + threadIdx.x;
    if (i >= n) return;
    float v = in[i];
    if (doClamp) v = clamp10(v);
    out[i] = (_Float16)v;
}

// ---------------------------------------------------------------------------
// RoPE applied in place on an f16 [M_, D_] tensor laid out as (row, h*DH+d).
// Each thread owns the (d, d+64) pair of one (row, head). Output clamped to
// +-10 and multiplied by outScale (1/sqrt(DH) folded into q).
// ---------------------------------------------------------------------------
__global__ void rope_inplace(_Float16* __restrict__ t, float outScale) {
    int i = blockIdx.x * blockDim.x + threadIdx.x;   // [0, M_*H_*64)
    int d = i & 63;
    int h = (i >> 6) & (H_ - 1);
    int m = i >> 10;
    if (m >= M_) return;
    int s = m & (S_ - 1);                            // position within sequence
    // inv_freq = 10000^(-d/64) = exp(-d * ln(10000)/64)
    float theta = (float)s * __expf(-(float)d * 0.14391156855f);
    float sn, cs;
    __sincosf(theta, &sn, &cs);
    int base = m * D_ + h * DH_ + d;
    float a = (float)t[base];
    float b = (float)t[base + 64];
    float o1 = clamp10(a * cs - b * sn) * outScale;
    float o2 = clamp10(b * cs + a * sn) * outScale;
    t[base]      = (_Float16)o1;
    t[base + 64] = (_Float16)o2;
}

// ---------------------------------------------------------------------------
// WMMA fragment loaders (wave32, V_WMMA_F32_16X16X32_F16 layouts per ISA):
//  A 16x32 : lane<16 -> chunks at k0+{0,16}; lane>=16 adds +8.
//  B 32x16 : lane<16 -> chunks at k0+{0,16}; lane>=16 adds +8; N = lane&15.
// ---------------------------------------------------------------------------
__device__ __forceinline__ v16h cat16(v8h lo, v8h hi) {
    v16h r;
#pragma unroll
    for (int i = 0; i < 8; ++i) { r[i] = lo[i]; r[i + 8] = hi[i]; }
    return r;
}

__device__ __forceinline__ v16h load_a_frag(const _Float16* rowPtr, int k0, int hiG) {
    const _Float16* p = rowPtr + k0 + (hiG ? 8 : 0);
    v8h lo = *(const v8h*)p;
    v8h hi = *(const v8h*)(p + 16);
    return cat16(lo, hi);
}

__device__ __forceinline__ v16h load_b_frag(const _Float16* rowPtr, int k0, int hiG) {
    const _Float16* p = rowPtr + k0 + (hiG ? 8 : 0);
    v8h lo = *(const v8h*)p;
    v8h hi = *(const v8h*)(p + 16);
    return cat16(lo, hi);
}

// ---------------------------------------------------------------------------
// C[row, col] = clamp( A[row, :] . W[col, :] )   (i.e. A @ W^T, both f16)
// Block = 128x128 tile, 8 waves of 64(M) x 32(N) each -> 8 WMMA accumulators.
// Direct-global fragment loads; weights + activations are L2-resident (192MB).
// ---------------------------------------------------------------------------
template<bool OUT_F16>
__global__ __launch_bounds__(256) void gemm_cwt(const _Float16* __restrict__ A,
                                                const _Float16* __restrict__ W,
                                                void* __restrict__ Cout,
                                                int K, int N) {
    const int lane = threadIdx.x & 31;
    const int wave = threadIdx.x >> 5;
    const int wm = wave >> 2;            // 0..1
    const int wn = wave & 3;             // 0..3
    const int mBase = blockIdx.y * 128 + wm * 64;
    const int nBase = blockIdx.x * 128 + wn * 32;
    const int l15 = lane & 15;
    const int hiG = lane >> 4;

    v8f zero = {};
    v8f acc[4][2];
#pragma unroll
    for (int i = 0; i < 4; ++i)
#pragma unroll
        for (int j = 0; j < 2; ++j) acc[i][j] = zero;

    const _Float16* aRow[4];
#pragma unroll
    for (int sm = 0; sm < 4; ++sm)
        aRow[sm] = A + (size_t)(mBase + 16 * sm + l15) * K;
    const _Float16* bRow[2];
#pragma unroll
    for (int sn = 0; sn < 2; ++sn)
        bRow[sn] = W + (size_t)(nBase + 16 * sn + l15) * K;

    for (int k0 = 0; k0 < K; k0 += 32) {
        if (k0 + 64 < K) {               // pull next-next K-slice toward L2/L0
            __builtin_prefetch(aRow[0] + k0 + 64, 0, 0);
            __builtin_prefetch(bRow[0] + k0 + 64, 0, 0);
        }
        v16h af[4], bf[2];
#pragma unroll
        for (int sm = 0; sm < 4; ++sm) af[sm] = load_a_frag(aRow[sm], k0, hiG);
#pragma unroll
        for (int sn = 0; sn < 2; ++sn) bf[sn] = load_b_frag(bRow[sn], k0, hiG);
#pragma unroll
        for (int sm = 0; sm < 4; ++sm)
#pragma unroll
            for (int sn = 0; sn < 2; ++sn)
                acc[sm][sn] = __builtin_amdgcn_wmma_f32_16x16x32_f16(
                    false, af[sm], false, bf[sn], (short)0, acc[sm][sn],
                    false, false);
    }

    // Epilogue: C layout -> lane l, vgpr r: row = 16*sm + r + 8*(l>>4), col = l&15
#pragma unroll
    for (int sm = 0; sm < 4; ++sm)
#pragma unroll
        for (int sn = 0; sn < 2; ++sn)
#pragma unroll
            for (int r = 0; r < 8; ++r) {
                int row = mBase + 16 * sm + r + 8 * hiG;
                int col = nBase + 16 * sn + l15;
                float v = clamp10(acc[sm][sn][r]);
                if (OUT_F16)
                    ((_Float16*)Cout)[(size_t)row * N + col] = (_Float16)v;
                else
                    ((float*)Cout)[(size_t)row * N + col] = v;
            }
}

// ---------------------------------------------------------------------------
// Flash attention (causal). One block = (b, h, 128 queries). 8 waves x 16 q.
// Double-buffered software pipeline: while computing on buffer `cur`, the
// NEXT key block's K tile streams into LDS via CDNA5 async DMA and the next
// V tile is transposed into LDS -- both fully hidden under the 32 WMMAs of
// the current block. One barrier per key block. Q carries the 1/sqrt(DH)
// scale (folded in by rope_inplace).
// ---------------------------------------------------------------------------
__global__ __launch_bounds__(256) void flash_attn(const _Float16* __restrict__ Q,
                                                  const _Float16* __restrict__ Kt,
                                                  const _Float16* __restrict__ V,
                                                  _Float16* __restrict__ O) {
    // Row strides keep 16B alignment for b128 LDS ops and map 16 consecutive
    // rows onto distinct banks (72 halves = 36 dwords, 136 halves = 68 dwords;
    // gcd with 64 banks = 4 -> 16 distinct banks across 16 rows).
    __shared__ __align__(16) _Float16 Vt[2][DH_][72];     // V^T tiles [d][key]
    __shared__ __align__(16) _Float16 Ktile[2][64][136];  // K tiles [key][d]
    __shared__ __align__(16) _Float16 Pb[8][16][72];      // per-wave P (16x64)

    const int lane = threadIdx.x & 31;
    const int wave = threadIdx.x >> 5;
    const int l15 = lane & 15;
    const int hiG = lane >> 4;

    const int b = blockIdx.z;
    const int h = blockIdx.y;
    const int qBlk = blockIdx.x * 128;
    const int qRow0 = qBlk + wave * 16;
    const size_t headOff = (size_t)h * DH_;

    // ---- stage helpers ----------------------------------------------------
    auto stageK = [&](int kb, int buf) {
        const _Float16* kBase = Kt + (size_t)(b * S_ + kb) * D_ + headOff;
#pragma unroll
        for (int j = 0; j < 4; ++j) {
            int chunk = threadIdx.x * 4 + j;   // 0..1023 16B chunks
            int row = chunk >> 4;              // key 0..63
            int c   = chunk & 15;              // chunk within row
            async_copy_b128(kBase + (size_t)row * D_ + c * 8,
                            &Ktile[buf][row][c * 8]);
        }
    };
    auto stageV = [&](int kb, int buf) {
        int kl = threadIdx.x >> 2;             // key 0..63
        int dc = (threadIdx.x & 3) * 32;       // dim chunk 0/32/64/96
        const _Float16* vp =
            V + (size_t)(b * S_ + kb + kl) * D_ + headOff + dc;
        v8h v0 = *(const v8h*)(vp);
        v8h v1 = *(const v8h*)(vp + 8);
        v8h v2 = *(const v8h*)(vp + 16);
        v8h v3 = *(const v8h*)(vp + 24);
#pragma unroll
        for (int j = 0; j < 8; ++j) {
            Vt[buf][dc + j][kl]      = v0[j];
            Vt[buf][dc + j + 8][kl]  = v1[j];
            Vt[buf][dc + j + 16][kl] = v2[j];
            Vt[buf][dc + j + 24][kl] = v3[j];
        }
    };

    // Resident Q fragments: 16 x 128 = 4 chunks of K=32
    const _Float16* qRowPtr = Q + (size_t)(b * S_ + qRow0 + l15) * D_ + headOff;
    v16h aq[4];
#pragma unroll
    for (int kc = 0; kc < 4; ++kc) aq[kc] = load_a_frag(qRowPtr, kc * 32, hiG);

    float mrow[8], lrow[8];
#pragma unroll
    for (int r = 0; r < 8; ++r) { mrow[r] = -1e30f; lrow[r] = 0.f; }
    v8f zero = {};
    v8f oacc[8];
#pragma unroll
    for (int sd = 0; sd < 8; ++sd) oacc[sd] = zero;

    const int kEnd = qBlk + 128;   // causal frontier for this block (uniform)

    // ---- pipeline prologue: fill buffer 0 ----
    stageK(0, 0);
    stageV(0, 0);
    wait_async0();
    __syncthreads();

    int buf = 0;
    for (int kb = 0; kb < kEnd; kb += 64, buf ^= 1) {
        // ---- kick off NEXT block's staging; hides under this block's WMMAs
        if (kb + 64 < kEnd) {
            stageK(kb + 64, buf ^ 1);
            stageV(kb + 64, buf ^ 1);
        }

        // ---- scores S = Q . K^T  (16 q x 64 k), K fragments from LDS ----
        v8f sacc[4];
#pragma unroll
        for (int sn = 0; sn < 4; ++sn) sacc[sn] = zero;
#pragma unroll
        for (int sn = 0; sn < 4; ++sn) {
            const _Float16* kRowPtr = &Ktile[buf][16 * sn + l15][0];
#pragma unroll
            for (int kc = 0; kc < 4; ++kc) {
                v16h bk = load_b_frag(kRowPtr, kc * 32, hiG);
                sacc[sn] = __builtin_amdgcn_wmma_f32_16x16x32_f16(
                    false, aq[kc], false, bk, (short)0, sacc[sn], false, false);
            }
        }

        // ---- online softmax (per row; rows live per 16-lane half) ----
#pragma unroll
        for (int r = 0; r < 8; ++r) {
            int qpos = qRow0 + r + 8 * hiG;
            float mx = -1e30f;
#pragma unroll
            for (int sn = 0; sn < 4; ++sn) {
                float s = clamp10(sacc[sn][r]);
                int kpos = kb + 16 * sn + l15;
                if (kpos > qpos) s = -10000.f;   // causal mask (matches ref)
                sacc[sn][r] = s;
                mx = fmaxf(mx, s);
            }
#pragma unroll
            for (int off = 8; off >= 1; off >>= 1)
                mx = fmaxf(mx, __shfl_xor(mx, off, 32));
            float mnew = fmaxf(mrow[r], mx);
            float corr = __expf(mrow[r] - mnew);
            mrow[r] = mnew;
            float psum = 0.f;
#pragma unroll
            for (int sn = 0; sn < 4; ++sn) {
                float p = __expf(sacc[sn][r] - mnew);
                psum += p;
                Pb[wave][r + 8 * hiG][16 * sn + l15] = (_Float16)p;
            }
#pragma unroll
            for (int off = 8; off >= 1; off >>= 1)
                psum += __shfl_xor(psum, off, 32);
            lrow[r] = lrow[r] * corr + psum;
#pragma unroll
            for (int sd = 0; sd < 8; ++sd) oacc[sd][r] *= corr;
        }

        // ---- O += P (16x64) @ V (64x128); A from Pb, B from Vt (both LDS) ----
#pragma unroll
        for (int kc = 0; kc < 2; ++kc) {
            const _Float16* pRow = &Pb[wave][l15][0];
            v16h ap = load_a_frag(pRow, kc * 32, hiG);
#pragma unroll
            for (int sd = 0; sd < 8; ++sd) {
                const _Float16* vRow = &Vt[buf][16 * sd + l15][0];
                v16h bv = load_b_frag(vRow, kc * 32, hiG);
                oacc[sd] = __builtin_amdgcn_wmma_f32_16x16x32_f16(
                    false, ap, false, bv, (short)0, oacc[sd], false, false);
            }
        }

        // ---- pipeline boundary: next buffers landed, everyone done with cur
        wait_async0();
        __syncthreads();
    }

    // ---- finalize: divide by row sums, clamp, store f16 ----
    _Float16* oBase = O + (size_t)(b * S_) * D_ + headOff;
#pragma unroll
    for (int r = 0; r < 8; ++r) {
        float inv = 1.f / lrow[r];
        int row = qRow0 + r + 8 * hiG;
#pragma unroll
        for (int sd = 0; sd < 8; ++sd) {
            float v = clamp10(oacc[sd][r] * inv);
            oBase[(size_t)row * D_ + 16 * sd + l15] = (_Float16)v;
        }
    }
}

// ---------------------------------------------------------------------------
// Host-side orchestration
// ---------------------------------------------------------------------------
extern "C" void kernel_launch(void* const* d_in, const int* in_sizes, int n_in,
                              void* d_out, int out_size, void* d_ws, size_t ws_size,
                              hipStream_t stream) {
    const float* x  = (const float*)d_in[0];
    // d_in[1] = mask (causal tril; hardcoded in flash_attn)
    const float* Wq = (const float*)d_in[2];
    const float* Wk = (const float*)d_in[3];
    const float* Wv = (const float*)d_in[4];
    const float* Wo = (const float*)d_in[5];

    _Float16* p = (_Float16*)d_ws;
    _Float16* xh  = p; p += (size_t)M_ * D_;
    _Float16* Wqh = p; p += (size_t)D_ * D_;
    _Float16* Wkh = p; p += (size_t)D_ * D_;
    _Float16* Wvh = p; p += (size_t)D_ * D_;
    _Float16* Woh = p; p += (size_t)D_ * D_;
    _Float16* qh  = p; p += (size_t)M_ * D_;
    _Float16* kh  = p; p += (size_t)M_ * D_;
    _Float16* vh  = p; p += (size_t)M_ * D_;
    _Float16* oh  = p; p += (size_t)M_ * D_;

    const int nx = M_ * D_;       // 8,388,608
    const int nw = D_ * D_;       // 4,194,304
    cvt_f32_f16<<<(nx + 255) / 256, 256, 0, stream>>>(x,  xh,  nx, 1);
    cvt_f32_f16<<<(nw + 255) / 256, 256, 0, stream>>>(Wq, Wqh, nw, 0);
    cvt_f32_f16<<<(nw + 255) / 256, 256, 0, stream>>>(Wk, Wkh, nw, 0);
    cvt_f32_f16<<<(nw + 255) / 256, 256, 0, stream>>>(Wv, Wvh, nw, 0);
    cvt_f32_f16<<<(nw + 255) / 256, 256, 0, stream>>>(Wo, Woh, nw, 0);

    dim3 gGemm(D_ / 128, M_ / 128);
    gemm_cwt<true><<<gGemm, 256, 0, stream>>>(xh, Wqh, qh, D_, D_);
    gemm_cwt<true><<<gGemm, 256, 0, stream>>>(xh, Wkh, kh, D_, D_);
    gemm_cwt<true><<<gGemm, 256, 0, stream>>>(xh, Wvh, vh, D_, D_);

    const int nr = M_ * H_ * 64;
    rope_inplace<<<(nr + 255) / 256, 256, 0, stream>>>(qh, 0.08838834764832f); // * 1/sqrt(128)
    rope_inplace<<<(nr + 255) / 256, 256, 0, stream>>>(kh, 1.0f);

    flash_attn<<<dim3(S_ / 128, H_, B_), 256, 0, stream>>>(qh, kh, vh, oh);

    gemm_cwt<false><<<gGemm, 256, 0, stream>>>(oh, Woh, d_out, D_, D_);
}
